// Equi_Nonlin_Grad_Module_71554155151895
// MI455X (gfx1250) — matrix operand order repaired and checked by reading, not compiled, not run on hardware
//
#include <hip/hip_runtime.h>
#include <hip/hip_bf16.h>

typedef __bf16 bf16_t;
typedef __attribute__((ext_vector_type(16))) __bf16 v16bf;
typedef __attribute__((ext_vector_type(8)))  float  v8f;

#define NS   50000
#define IRR  240

// ---- forward: symmetric (upper-triangle) dot features, padded to 32-chunks ----
// l=0: 64*65/2 = 2080 (65 chunks, exact)   l=1: 32*33/2 = 528 -> 544 (17 chunks)
// l=2: 16*17/2 = 136 -> 160 (5 chunks)     total padded NFP = 2784 = 87*32
#define FCH0 65
#define FCH1 17
#define FCH2 5
#define FCHT (FCH0+FCH1+FCH2)     // 87
#define NFP  (FCHT*32)            // 2784
#define L1_BASE  2080
#define L1_VALID 2608             // 2080+528
#define L2_BASE  2624
#define L2_VALID 2760             // 2624+136

#define BCH  336                  // backward feature chunks of 16 (full 5376)

#define WCAT_FRAGS (FCHT*4*32)    // 11136 fragments (v16bf, 32B each)
#define WSYM_FRAGS (BCH*2*32)     // 21504
#define MLP_FRAGS  (2*4*32)       // 256 per MLP weight set
#define OFF_WCAT 0
#define OFF_WSYM (WCAT_FRAGS*32)                 // 356352
#define OFF_MLP  (OFF_WSYM + WSYM_FRAGS*32)      // 1044480
#define OFF_C3   (OFF_MLP + 5*MLP_FRAGS*32)      // 1085440
#define OFF_DESC (OFF_C3 + 256)                  // 1085696
// total ws bytes needed: 1085696 + 2784*4 = 1096832 (~1.05 MB)

__device__ __forceinline__ void f2luv(int f, int& l, int& u, int& v)
{
    if (f < 4096)      { l = 0; u = f >> 6; v = f & 63; }
    else if (f < 5120) { l = 1; int g = f - 4096; u = g >> 5; v = g & 31; }
    else               { l = 2; int g = f - 5120; u = g >> 4; v = g & 15; }
}

// prep pass 1: backward Wsym fragments, MLP weight fragments, c3, and the
// triangular feature descriptor table (packed bu | bv<<16 x-offsets).
__global__ void prep1_kernel(const float* __restrict__ W0, const float* __restrict__ W1,
                             const float* __restrict__ W2, const float* __restrict__ A1,
                             const float* __restrict__ A2, const float* __restrict__ A3,
                             unsigned char* __restrict__ ws)
{
    const int tid = blockIdx.x * blockDim.x + threadIdx.x;
    const float c0 = rsqrtf(5376.0f);                 // PATH_NORM
    const float cl[3] = { c0, c0 * rsqrtf(3.0f), c0 * rsqrtf(5.0f) };

    bf16_t*   wsym = (bf16_t*)  (ws + OFF_WSYM);
    bf16_t*   mlp  = (bf16_t*)  (ws + OFF_MLP);
    float*    c3   = (float*)   (ws + OFF_C3);
    unsigned* desc = (unsigned*)(ws + OFF_DESC);
    const int NWS = WSYM_FRAGS * 16;                  // 344064

    if (tid < NWS) {
        // Wsym[f,k] = c_l*(W[u,v,k]+W[v,u,k]): B frag (fchunk fb, K-chunk kc)
        int e = tid & 15, lane = (tid >> 4) & 31, kc = (tid >> 9) & 1, fb = tid >> 10;
        int f = fb * 16 + (lane & 15);
        int k = kc * 32 + ((e < 8) ? (8 * (lane >> 4) + e) : (16 + 8 * (lane >> 4) + (e - 8)));
        int l, u, v; f2luv(f, l, u, v);
        float val;
        if (l == 0)      val = W0[((u << 6) + v) * 64 + k] + W0[((v << 6) + u) * 64 + k];
        else if (l == 1) val = W1[((u << 5) + v) * 64 + k] + W1[((v << 5) + u) * 64 + k];
        else             val = W2[((u << 4) + v) * 64 + k] + W2[((v << 4) + u) * 64 + k];
        wsym[tid] = (bf16_t)(cl[l] * val);
    } else if (tid < NWS + 5 * MLP_FRAGS * 16) {
        // sets 0..2 = A1^T,A2^T,A3^T (forward y=h@A^T); 3..4 = A1,A2 (backward)
        int i = tid - NWS;
        int set = i >> 12;
        int j = i & 4095;
        int e = j & 15, lane = (j >> 4) & 31, t = (j >> 9) & 3, kc = j >> 11;
        int ncol = 16 * t + (lane & 15);
        int krow = kc * 32 + ((e < 8) ? (8 * (lane >> 4) + e) : (16 + 8 * (lane >> 4) + (e - 8)));
        const float* A = (set == 0 || set == 3) ? A1 : (set == 1 || set == 4) ? A2 : A3;
        float val = (set < 3) ? A[ncol * 64 + krow] : A[krow * 64 + ncol];
        mlp[i] = (bf16_t)val;
    } else if (tid < NWS + 5 * MLP_FRAGS * 16 + 64) {
        int j = tid - (NWS + 5 * MLP_FRAGS * 16);
        float s = 0.f;
        for (int k = 0; k < 64; ++k) s += A3[k * 64 + j];
        c3[j] = s;
    } else if (tid < NWS + 5 * MLP_FRAGS * 16 + 64 + 152) {
        int r = tid - (NWS + 5 * MLP_FRAGS * 16 + 64);
        if (r < 64) {                      // l=0 row u: x offsets bu = u
            int u = r;
            int fp = u * 64 - (u * (u - 1)) / 2;
            for (int v = u; v < 64; ++v) desc[fp++] = (unsigned)u | ((unsigned)v << 16);
        } else if (r < 96) {               // l=1 row u: bu = 64 + 3u
            int u = r - 64;
            int fp = L1_BASE + u * 32 - (u * (u - 1)) / 2;
            for (int v = u; v < 32; ++v)
                desc[fp++] = (unsigned)(64 + u * 3) | ((unsigned)(64 + v * 3) << 16);
        } else if (r < 112) {              // l=2 row u: bu = 160 + 5u
            int u = r - 96;
            int fp = L2_BASE + u * 16 - (u * (u - 1)) / 2;
            for (int v = u; v < 16; ++v)
                desc[fp++] = (unsigned)(160 + u * 5) | ((unsigned)(160 + v * 5) << 16);
        } else {                           // 40 padding slots -> desc 0 (weight also 0)
            int p = r - 112;
            int fp = (p < 16) ? (L1_VALID + p) : (L2_VALID + (p - 16));
            desc[fp] = 0u;
        }
    }
}

// prep pass 2: forward Wtri fragments (consumes desc); diagonal gets W[u,u],
// off-diagonal gets W[u,v]+W[v,u]; padded features get 0.
__global__ void prep2_kernel(const float* __restrict__ W0, const float* __restrict__ W1,
                             const float* __restrict__ W2, unsigned char* __restrict__ ws)
{
    const int tid = blockIdx.x * blockDim.x + threadIdx.x;
    if (tid >= WCAT_FRAGS * 16) return;
    const float c0 = rsqrtf(5376.0f);
    bf16_t*         wcat = (bf16_t*)        (ws + OFF_WCAT);
    const unsigned* desc = (const unsigned*)(ws + OFF_DESC);

    int e = tid & 15, lane = (tid >> 4) & 31, t = (tid >> 9) & 3, c = tid >> 11;
    int k  = 16 * t + (lane & 15);
    int fp = c * 32 + ((e < 8) ? (8 * (lane >> 4) + e) : (16 + 8 * (lane >> 4) + (e - 8)));

    float val = 0.f;
    if (c < FCH0) {
        unsigned d = desc[fp];
        int u = d & 0xffff, v = d >> 16;
        float s = W0[((u << 6) + v) * 64 + k];
        if (u != v) s += W0[((v << 6) + u) * 64 + k];
        val = c0 * s;
    } else if (c < FCH0 + FCH1) {
        if (fp < L1_VALID) {
            unsigned d = desc[fp];
            int u = ((int)(d & 0xffff) - 64) / 3, v = ((int)(d >> 16) - 64) / 3;
            float s = W1[((u << 5) + v) * 64 + k];
            if (u != v) s += W1[((v << 5) + u) * 64 + k];
            val = c0 * rsqrtf(3.0f) * s;
        }
    } else {
        if (fp < L2_VALID) {
            unsigned d = desc[fp];
            int u = ((int)(d & 0xffff) - 160) / 5, v = ((int)(d >> 16) - 160) / 5;
            float s = W2[((u << 4) + v) * 64 + k];
            if (u != v) s += W2[((v << 4) + u) * 64 + k];
            val = c0 * rsqrtf(5.0f) * s;
        }
    }
    wcat[tid] = (bf16_t)val;
}

__device__ __forceinline__ v8f wmma_bf16(v16bf a, v16bf b, v8f c)
{
    return __builtin_amdgcn_wmma_f32_16x16x32_bf16(false, a, false, b, (short)0, c, false, false);
}

__device__ __forceinline__ void load_afrags(const float (*tile)[64], int lo, int ab,
                                            v16bf& a0, v16bf& a1)
{
#pragma unroll
    for (int e = 0; e < 16; ++e) {
        int kk = (e < 8) ? (ab + e) : (16 + ab + (e - 8));
        a0[e] = (bf16_t)tile[lo][kk];
        a1[e] = (bf16_t)tile[lo][32 + kk];
    }
}

__device__ __forceinline__ void gemm64(v16bf a0, v16bf a1, const v16bf* B, int lane,
                                       v8f (&acc)[4])
{
#pragma unroll
    for (int t = 0; t < 4; ++t) {
        v8f c = {};
        c = wmma_bf16(a0, B[(0 * 4 + t) * 32 + lane], c);
        c = wmma_bf16(a1, B[(1 * 4 + t) * 32 + lane], c);
        acc[t] = c;
    }
}

__global__ __launch_bounds__(64)
void fused_fwd_bwd(const float* __restrict__ tin,
                   const unsigned char* __restrict__ ws,
                   const float* __restrict__ b1, const float* __restrict__ b2,
                   const float* __restrict__ b3, float* __restrict__ out)
{
    __shared__ float xs [2][16][IRR];   // input irrep tile per wave
    __shared__ float ht [2][16][64];    // current hidden tile (h0/h1/h2/dz1/dh0)
    __shared__ float sp1[2][16][64];    // silu'(z1)
    __shared__ float sp2[2][16][64];    // silu'(z2)
    __shared__ float gt [2][16][16];    // backward G chunk
    __shared__ float c3s[64];

    const int tid  = threadIdx.x;
    const int w    = tid >> 5;
    const int lane = tid & 31;
    const int lo   = lane & 15;
    const int hi   = lane >> 4;
    const int ab   = 8 * hi;            // C/D row base & A K base for this half-wave
    const int s0   = blockIdx.x * 32 + w * 16;

    const v16bf*    WCAT  = (const v16bf*)   (ws + OFF_WCAT);
    const v16bf*    WSYM  = (const v16bf*)   (ws + OFF_WSYM);
    const v16bf*    B1f   = (const v16bf*)   (ws + OFF_MLP + 0 * MLP_FRAGS * 32);
    const v16bf*    B2f   = (const v16bf*)   (ws + OFF_MLP + 1 * MLP_FRAGS * 32);
    const v16bf*    B3f   = (const v16bf*)   (ws + OFF_MLP + 2 * MLP_FRAGS * 32);
    const v16bf*    B1b   = (const v16bf*)   (ws + OFF_MLP + 3 * MLP_FRAGS * 32);
    const v16bf*    B2b   = (const v16bf*)   (ws + OFF_MLP + 4 * MLP_FRAGS * 32);
    const float*    c3g   = (const float*)   (ws + OFF_C3);
    const unsigned* descg = (const unsigned*)(ws + OFF_DESC);

    c3s[tid] = c3g[tid];                // 64 threads cover 64 entries

    // ---- stage x tile with CDNA5 async global->LDS (no VGPR round-trip) ----
    {
        unsigned lds_base = (unsigned)(size_t)(&xs[w][0][0]);    // low 32 bits = LDS offset
        unsigned long long gbase = (unsigned long long)(size_t)tin;
        for (int i = lane; i < 960; i += 32) {                   // 960 x 16B per wave tile
            int s = i / 60, c16 = i - s * 60;                    // 60 x 16B per 240-f32 row
            int g = s0 + s; g = g < NS ? g : NS - 1;             // clamp keeps lanes active
            unsigned goff = (unsigned)(g * 960 + c16 * 16);
            unsigned loff = lds_base + (unsigned)(s * 960 + c16 * 16);
            asm volatile("global_load_async_to_lds_b128 %0, %1, %2"
                         :: "v"(loff), "v"(goff), "s"(gbase) : "memory");
        }
        asm volatile("s_wait_asynccnt 0" ::: "memory");
    }
    __syncthreads();   // cross-wave dep is c3s; also publishes async LDS writes

    const float* xrow = &xs[w][lo][0];  // this lane's A-row sample

    // ------- forward FCTP (triangular): h0 = Dtri(2784) @ Wtri -------
    v8f acc0 = {}, acc1 = {}, acc2 = {}, acc3 = {};

    for (int c = 0; c < FCH0; ++c) {                         // l=0: D = x[u]*x[v]
        const unsigned* dc = descg + c * 32;
        v16bf af;
#pragma unroll
        for (int e = 0; e < 16; ++e) {
            int kk = (e < 8) ? (ab + e) : (16 + ab + (e - 8));
            unsigned d = dc[kk];
            af[e] = (bf16_t)(xrow[d & 0xffff] * xrow[d >> 16]);
        }
        acc0 = wmma_bf16(af, WCAT[(c * 4 + 0) * 32 + lane], acc0);
        acc1 = wmma_bf16(af, WCAT[(c * 4 + 1) * 32 + lane], acc1);
        acc2 = wmma_bf16(af, WCAT[(c * 4 + 2) * 32 + lane], acc2);
        acc3 = wmma_bf16(af, WCAT[(c * 4 + 3) * 32 + lane], acc3);
    }
    for (int c = FCH0; c < FCH0 + FCH1; ++c) {               // l=1: 3-term dot
        const unsigned* dc = descg + c * 32;
        v16bf af;
#pragma unroll
        for (int e = 0; e < 16; ++e) {
            int kk = (e < 8) ? (ab + e) : (16 + ab + (e - 8));
            unsigned d = dc[kk];
            const float* xa = xrow + (d & 0xffff);
            const float* xb = xrow + (d >> 16);
            af[e] = (bf16_t)(xa[0]*xb[0] + xa[1]*xb[1] + xa[2]*xb[2]);
        }
        acc0 = wmma_bf16(af, WCAT[(c * 4 + 0) * 32 + lane], acc0);
        acc1 = wmma_bf16(af, WCAT[(c * 4 + 1) * 32 + lane], acc1);
        acc2 = wmma_bf16(af, WCAT[(c * 4 + 2) * 32 + lane], acc2);
        acc3 = wmma_bf16(af, WCAT[(c * 4 + 3) * 32 + lane], acc3);
    }
    for (int c = FCH0 + FCH1; c < FCHT; ++c) {               // l=2: 5-term dot
        const unsigned* dc = descg + c * 32;
        v16bf af;
#pragma unroll
        for (int e = 0; e < 16; ++e) {
            int kk = (e < 8) ? (ab + e) : (16 + ab + (e - 8));
            unsigned d = dc[kk];
            const float* xa = xrow + (d & 0xffff);
            const float* xb = xrow + (d >> 16);
            af[e] = (bf16_t)(xa[0]*xb[0] + xa[1]*xb[1] + xa[2]*xb[2] + xa[3]*xb[3] + xa[4]*xb[4]);
        }
        acc0 = wmma_bf16(af, WCAT[(c * 4 + 0) * 32 + lane], acc0);
        acc1 = wmma_bf16(af, WCAT[(c * 4 + 1) * 32 + lane], acc1);
        acc2 = wmma_bf16(af, WCAT[(c * 4 + 2) * 32 + lane], acc2);
        acc3 = wmma_bf16(af, WCAT[(c * 4 + 3) * 32 + lane], acc3);
    }

    // h0 -> LDS tile (C/D layout: element r -> row ab+r, col 16t+lo)
#pragma unroll
    for (int r = 0; r < 8; ++r) {
        ht[w][ab + r][ 0 + lo] = acc0[r];
        ht[w][ab + r][16 + lo] = acc1[r];
        ht[w][ab + r][32 + lo] = acc2[r];
        ht[w][ab + r][48 + lo] = acc3[r];
    }

    // ---------------- forward MLP ----------------
    v16bf a0, a1;
    v8f   z[4];

    load_afrags(ht[w], lo, ab, a0, a1);            // layer 1
    gemm64(a0, a1, B1f, lane, z);
#pragma unroll
    for (int t = 0; t < 4; ++t) {
        float bv = b1[16 * t + lo];
#pragma unroll
        for (int r = 0; r < 8; ++r) {
            float zz  = z[t][r] + bv;
            float sig = 1.0f / (1.0f + __expf(-zz));
            ht [w][ab + r][16 * t + lo] = zz * sig;
            sp1[w][ab + r][16 * t + lo] = sig * (1.0f + zz * (1.0f - sig));
        }
    }
    load_afrags(ht[w], lo, ab, a0, a1);            // layer 2
    gemm64(a0, a1, B2f, lane, z);
#pragma unroll
    for (int t = 0; t < 4; ++t) {
        float bv = b2[16 * t + lo];
#pragma unroll
        for (int r = 0; r < 8; ++r) {
            float zz  = z[t][r] + bv;
            float sig = 1.0f / (1.0f + __expf(-zz));
            ht [w][ab + r][16 * t + lo] = zz * sig;
            sp2[w][ab + r][16 * t + lo] = sig * (1.0f + zz * (1.0f - sig));
        }
    }
    load_afrags(ht[w], lo, ab, a0, a1);            // layer 3 -> x output
    gemm64(a0, a1, B3f, lane, z);
#pragma unroll
    for (int t = 0; t < 4; ++t) {
        float bv = b3[16 * t + lo];
#pragma unroll
        for (int r = 0; r < 8; ++r) {
            int smp = s0 + ab + r;
            if (smp < NS) out[(size_t)smp * 64 + 16 * t + lo] = z[t][r] + bv;
        }
    }

    // ---------------- backward MLP (cotangent = ones) ----------------
#pragma unroll
    for (int e = 0; e < 16; ++e) {                 // dz2 = c3 * sp2
        int kk = (e < 8) ? (ab + e) : (16 + ab + (e - 8));
        a0[e] = (bf16_t)(sp2[w][lo][kk]      * c3s[kk]);
        a1[e] = (bf16_t)(sp2[w][lo][32 + kk] * c3s[32 + kk]);
    }
    gemm64(a0, a1, B2b, lane, z);                  // dh1 = dz2 @ A2
#pragma unroll
    for (int t = 0; t < 4; ++t)
#pragma unroll
        for (int r = 0; r < 8; ++r)                // dz1 = dh1 * sp1
            ht[w][ab + r][16 * t + lo] = z[t][r] * sp1[w][ab + r][16 * t + lo];

    load_afrags(ht[w], lo, ab, a0, a1);
    gemm64(a0, a1, B1b, lane, z);                  // dh0 = dz1 @ A1
#pragma unroll
    for (int t = 0; t < 4; ++t)
#pragma unroll
        for (int r = 0; r < 8; ++r)
            ht[w][ab + r][16 * t + lo] = z[t][r];

    v16bf da0, da1;                                // dh0 A-frags, reused for 336 chunks
    load_afrags(ht[w], lo, ab, da0, da1);

    // -------- backward FCTP: G = dh0 @ Wsym ; grad = G x --------
    const size_t ybase = (size_t)NS * 64;
    const int smp_ok = (s0 + lo) < NS;

    for (int u = 0; u < 64; ++u) {                 // l=0
        float acc = 0.f;
        for (int seg = 0; seg < 4; ++seg) {
            int fb = u * 4 + seg;
            v8f g = {};
            g = wmma_bf16(da0, WSYM[(fb * 2 + 0) * 32 + lane], g);
            g = wmma_bf16(da1, WSYM[(fb * 2 + 1) * 32 + lane], g);
#pragma unroll
            for (int r = 0; r < 8; ++r) gt[w][ab + r][lo] = g[r];
            if (lane < 16) {                       // wave32 lockstep + in-order LDS
                const float* xr = &xs[w][lane][0];
                int v0 = seg * 16;
#pragma unroll
                for (int j = 0; j < 16; ++j) acc += gt[w][lane][j] * xr[v0 + j];
            }
        }
        if (lane < 16 && smp_ok) out[ybase + (size_t)(s0 + lane) * IRR + u] = acc;
    }
    for (int u = 0; u < 32; ++u) {                 // l=1
        float g0 = 0.f, g1 = 0.f, g2 = 0.f;
        for (int seg = 0; seg < 2; ++seg) {
            int fb = 256 + u * 2 + seg;
            v8f g = {};
            g = wmma_bf16(da0, WSYM[(fb * 2 + 0) * 32 + lane], g);
            g = wmma_bf16(da1, WSYM[(fb * 2 + 1) * 32 + lane], g);
#pragma unroll
            for (int r = 0; r < 8; ++r) gt[w][ab + r][lo] = g[r];
            if (lane < 16) {
                int v0 = seg * 16;
#pragma unroll
                for (int j = 0; j < 16; ++j) {
                    float gv = gt[w][lane][j];
                    const float* xv = &xs[w][lane][64 + (v0 + j) * 3];
                    g0 += gv * xv[0]; g1 += gv * xv[1]; g2 += gv * xv[2];
                }
            }
        }
        if (lane < 16 && smp_ok) {
            size_t b = ybase + (size_t)(s0 + lane) * IRR + 64 + u * 3;
            out[b] = g0; out[b + 1] = g1; out[b + 2] = g2;
        }
    }
    for (int u = 0; u < 16; ++u) {                 // l=2
        int fb = 320 + u;
        v8f g = {};
        g = wmma_bf16(da0, WSYM[(fb * 2 + 0) * 32 + lane], g);
        g = wmma_bf16(da1, WSYM[(fb * 2 + 1) * 32 + lane], g);
#pragma unroll
        for (int r = 0; r < 8; ++r) gt[w][ab + r][lo] = g[r];
        float g5[5] = {0.f, 0.f, 0.f, 0.f, 0.f};
        if (lane < 16) {
#pragma unroll
            for (int j = 0; j < 16; ++j) {
                float gv = gt[w][lane][j];
                const float* xv = &xs[w][lane][160 + j * 5];
#pragma unroll
                for (int m = 0; m < 5; ++m) g5[m] += gv * xv[m];
            }
        }
        if (lane < 16 && smp_ok) {
            size_t b = ybase + (size_t)(s0 + lane) * IRR + 160 + u * 5;
#pragma unroll
            for (int m = 0; m < 5; ++m) out[b + m] = g5[m];
        }
    }
}

extern "C" void kernel_launch(void* const* d_in, const int* in_sizes, int n_in,
                              void* d_out, int out_size, void* d_ws, size_t ws_size,
                              hipStream_t stream)
{
    (void)in_sizes; (void)n_in; (void)out_size; (void)ws_size;
    const float* tin = (const float*)d_in[0];
    const float* W0  = (const float*)d_in[1];
    const float* W1  = (const float*)d_in[2];
    const float* W2  = (const float*)d_in[3];
    const float* A1  = (const float*)d_in[4];
    const float* b1  = (const float*)d_in[5];
    const float* A2  = (const float*)d_in[6];
    const float* b2  = (const float*)d_in[7];
    const float* A3  = (const float*)d_in[8];
    const float* b3  = (const float*)d_in[9];
    float* out = (float*)d_out;
    unsigned char* ws = (unsigned char*)d_ws;

    const int p1_threads = WSYM_FRAGS * 16 + 5 * MLP_FRAGS * 16 + 64 + 152;
    prep1_kernel<<<(p1_threads + 255) / 256, 256, 0, stream>>>(W0, W1, W2, A1, A2, A3, ws);
    const int p2_threads = WCAT_FRAGS * 16;
    prep2_kernel<<<(p2_threads + 255) / 256, 256, 0, stream>>>(W0, W1, W2, ws);

    const int blocks = (NS + 31) / 32;   // 32 samples per block (2 waves x 16)
    fused_fwd_bwd<<<blocks, 64, 0, stream>>>(tin, ws, b1, b2, b3, out);
}
